// SinkhornLoss_59854664237170
// MI455X (gfx1250) — matrix operand order repaired and checked by reading, not compiled, not run on hardware
//
#include <hip/hip_runtime.h>

#define N     2304          // 48*48
#define BATCH 32
#define INV_EPS 100.0f
#define EPSF    0.01f

typedef __attribute__((ext_vector_type(16))) __bf16 v16bf;
typedef __attribute__((ext_vector_type(8)))  __bf16 v8bf;
typedef __attribute__((ext_vector_type(8)))  float  v8f;

// ---------------------------------------------------------------- reductions
__device__ __forceinline__ float blockReduceSum(float val) {
    __shared__ float sh[8];
    __syncthreads();                       // safe for repeated calls
    for (int o = 16; o > 0; o >>= 1) val += __shfl_xor(val, o, 32);
    const int lane = threadIdx.x & 31, wid = threadIdx.x >> 5;
    if (lane == 0) sh[wid] = val;
    __syncthreads();
    val = (threadIdx.x < (blockDim.x >> 5)) ? sh[threadIdx.x] : 0.0f;
    if (wid == 0)
        for (int o = 4; o > 0; o >>= 1) val += __shfl_xor(val, o, 32);
    return val;                            // valid on thread 0
}

// ------------------------------------------------- A = exp(-C/eps), G = C*A
__global__ void precomp_kernel(const float* __restrict__ C,
                               __bf16* __restrict__ A, __bf16* __restrict__ G) {
    int idx = blockIdx.x * blockDim.x + threadIdx.x;
    float c = C[idx];
    float a = __expf(-c * INV_EPS);
    A[idx] = (__bf16)a;
    G[idx] = (__bf16)(a * c);
}

// ------------------------------- per-batch normalization, log masses, u=v=0
// layout: u, v, lu, lv stored as [i*32 + b]  (row i, batch b)
__global__ void init_kernel(const float* __restrict__ pred,
                            const float* __restrict__ targ,
                            float* __restrict__ u,  float* __restrict__ v,
                            float* __restrict__ lu, float* __restrict__ lv) {
    const int b = blockIdx.x;
    const float* p = pred + (size_t)b * 3 * N;   // channel 0
    const float* t = targ + (size_t)b * 3 * N;
    float sp = 0.f, st = 0.f;
    for (int i = threadIdx.x; i < N; i += blockDim.x) {
        sp += p[i] + 1e-9f;
        st += t[i] + 1e-9f;
    }
    sp = blockReduceSum(sp);
    st = blockReduceSum(st);
    __shared__ float shp, sht;
    if (threadIdx.x == 0) { shp = __logf(sp); sht = __logf(st); }
    __syncthreads();
    const float lsp = shp, lst = sht;
    for (int i = threadIdx.x; i < N; i += blockDim.x) {
        lu[i * 32 + b] = __logf(p[i] + 1e-9f) - lsp;
        lv[i * 32 + b] = __logf(t[i] + 1e-9f) - lst;
        u[i * 32 + b] = 0.f;
        v[i * 32 + b] = 0.f;
    }
}

// -------------------------- Xt[b][k] = bf16(exp(x[k*32+b]/eps))  (B operand)
__global__ void expT_kernel(const float* __restrict__ x, __bf16* __restrict__ Xt) {
    const int b = blockIdx.y;
    const int i = blockIdx.x * blockDim.x + threadIdx.x;
    if (i < N) Xt[(size_t)b * N + i] = (__bf16)__expf(x[i * 32 + b] * INV_EPS);
}

// ----------------------------------------------------------- WMMA GEMM core
// S[i][b] = sum_k Amat[i][k] * Bt[b][k]   (Bt is the transposed B operand)
// Each wave computes a 32-row x 32-col strip: 2 row tiles x 2 col tiles,
// reusing the B fragments across both row tiles (halves B-side L2 traffic,
// gives 4 independent WMMA accumulation chains for latency hiding).
// doUpdate==1: x[i*32+b] = eps*(logx - log(exp(x/eps)*S + 1e-6)) + x
// doUpdate==0: Sout[i*32+b] = S
__global__ void __launch_bounds__(128)
gemm_kernel(const __bf16* __restrict__ Amat, const __bf16* __restrict__ Bt,
            float* __restrict__ x, const float* __restrict__ logx,
            float* __restrict__ Sout, int doUpdate) {
    const int wave = threadIdx.x >> 5;
    const int lane = threadIdx.x & 31;
    const int row0 = (blockIdx.x * 4 + wave) * 32;   // 72 strip tiles total
    const int kg   = lane >> 4;                      // lane group (0/1)
    const int ncol = lane & 15;

    // A fragment rows: lane holds matrix row (rowbase + ncol),
//   values[0..7]  = K = k0 + kg*16 + 0..7
//   values[8..15] = K = k0 + kg*16 + 8 + 0..7
    const __bf16* a0 = Amat + (size_t)(row0 + ncol)      * N;
    const __bf16* a1 = Amat + (size_t)(row0 + 16 + ncol) * N;
    // B fragment: lane holds column ncol (+tile*16), K = k0 + kg*16 + 0..15
    const __bf16* b0 = Bt + (size_t)ncol        * N + kg * 16;
    const __bf16* b1 = Bt + (size_t)(16 + ncol) * N + kg * 16;

    v8f acc00 = {}; v8f acc01 = {};   // row tile 0, col tiles 0/1
    v8f acc10 = {}; v8f acc11 = {};   // row tile 1, col tiles 0/1

    for (int k0 = 0; k0 < N; k0 += 32) {
        v8bf a0lo = *(const v8bf*)(a0 + k0 + kg * 16);
        v8bf a0hi = *(const v8bf*)(a0 + k0 + 8 + kg * 16);
        v8bf a1lo = *(const v8bf*)(a1 + k0 + kg * 16);
        v8bf a1hi = *(const v8bf*)(a1 + k0 + 8 + kg * 16);
        v16bf af0, af1;
#pragma unroll
        for (int t = 0; t < 8; ++t) {
            af0[t] = a0lo[t]; af0[t + 8] = a0hi[t];
            af1[t] = a1lo[t]; af1[t + 8] = a1hi[t];
        }
        v16bf bf0 = *(const v16bf*)(b0 + k0);
        v16bf bf1 = *(const v16bf*)(b1 + k0);
        acc00 = __builtin_amdgcn_wmma_f32_16x16x32_bf16(
                    false, af0, false, bf0, (short)0, acc00, false, false);
        acc01 = __builtin_amdgcn_wmma_f32_16x16x32_bf16(
                    false, af0, false, bf1, (short)0, acc01, false, false);
        acc10 = __builtin_amdgcn_wmma_f32_16x16x32_bf16(
                    false, af1, false, bf0, (short)0, acc10, false, false);
        acc11 = __builtin_amdgcn_wmma_f32_16x16x32_bf16(
                    false, af1, false, bf1, (short)0, acc11, false, false);
    }

    // C/D layout: lanes 0-15 -> M=r, N=lane ; lanes 16-31 -> M=8+r, N=lane-16
#pragma unroll
    for (int mt = 0; mt < 2; ++mt) {
        const v8f* accA = mt ? &acc10 : &acc00;
        const v8f* accB = mt ? &acc11 : &acc01;
#pragma unroll
        for (int r = 0; r < 8; ++r) {
            const int i = row0 + mt * 16 + r + 8 * kg;
            {
                const int idx = i * 32 + ncol;
                const float S = (*accA)[r];
                if (doUpdate) {
                    const float xv = x[idx];
                    x[idx] = EPSF * (logx[idx] -
                             __logf(__expf(xv * INV_EPS) * S + 1e-6f)) + xv;
                } else {
                    Sout[idx] = S;
                }
            }
            {
                const int idx = i * 32 + 16 + ncol;
                const float S = (*accB)[r];
                if (doUpdate) {
                    const float xv = x[idx];
                    x[idx] = EPSF * (logx[idx] -
                             __logf(__expf(xv * INV_EPS) * S + 1e-6f)) + xv;
                } else {
                    Sout[idx] = S;
                }
            }
        }
    }
}

// --------------------------------- cost[b] = sum_i exp(u[i,b]/eps) * S[i,b]
__global__ void final_kernel(const float* __restrict__ S,
                             const float* __restrict__ u,
                             float* __restrict__ out) {
    const int b = blockIdx.x;
    float acc = 0.f;
    for (int i = threadIdx.x; i < N; i += blockDim.x)
        acc += __expf(u[i * 32 + b] * INV_EPS) * S[i * 32 + b];
    acc = blockReduceSum(acc);
    if (threadIdx.x == 0) out[b] = acc;
}

// --------------------------------------------------------------------- host
extern "C" void kernel_launch(void* const* d_in, const int* in_sizes, int n_in,
                              void* d_out, int out_size, void* d_ws, size_t ws_size,
                              hipStream_t stream) {
    const float* img_pred   = (const float*)d_in[0];   // [32,3,48,48]
    const float* img_target = (const float*)d_in[1];   // [32,3,48,48]
    const float* cost       = (const float*)d_in[2];   // [2304,2304]
    float* out = (float*)d_out;                        // [32]

    char* wsb = (char*)d_ws;
    size_t off = 0;
    auto carve = [&](size_t bytes) -> void* {
        void* p = wsb + off;
        off += (bytes + 255) & ~(size_t)255;
        return p;
    };
    __bf16* A  = (__bf16*)carve((size_t)N * N * sizeof(__bf16));   // exp(-C/eps)
    __bf16* G  = (__bf16*)carve((size_t)N * N * sizeof(__bf16));   // C*exp(-C/eps)
    __bf16* Wt = (__bf16*)carve((size_t)BATCH * N * sizeof(__bf16));
    __bf16* Ut = (__bf16*)carve((size_t)BATCH * N * sizeof(__bf16));
    float* u  = (float*)carve((size_t)N * BATCH * sizeof(float));
    float* v  = (float*)carve((size_t)N * BATCH * sizeof(float));
    float* lu = (float*)carve((size_t)N * BATCH * sizeof(float));
    float* lv = (float*)carve((size_t)N * BATCH * sizeof(float));
    float* S  = (float*)carve((size_t)N * BATCH * sizeof(float));

    precomp_kernel<<<(N * N) / 256, 256, 0, stream>>>(cost, A, G);
    init_kernel<<<BATCH, 256, 0, stream>>>(img_pred, img_target, u, v, lu, lv);

    dim3 egrid((N + 255) / 256, BATCH);
    const int ggrid = N / 32 / 4;   // 18 blocks x 4 waves = 72 strip tiles

    for (int it = 0; it < 5; ++it) {
        // u <- eps*(log_u - log(exp(u/eps)*(A@exp(v/eps)) + 1e-6)) + u
        expT_kernel<<<egrid, 256, 0, stream>>>(v, Wt);
        gemm_kernel<<<ggrid, 128, 0, stream>>>(A, Wt, u, lu, S, 1);
        // v <- eps*(log_v - log(exp(v/eps)*(A@exp(u/eps)) + 1e-6)) + v   (A symmetric)
        expT_kernel<<<egrid, 256, 0, stream>>>(u, Ut);
        gemm_kernel<<<ggrid, 128, 0, stream>>>(A, Ut, v, lv, S, 1);
    }

    // cost[b] = sum_i exp(u_i/eps) * (G @ exp(v/eps))_i
    expT_kernel<<<egrid, 256, 0, stream>>>(v, Wt);
    gemm_kernel<<<ggrid, 128, 0, stream>>>(G, Wt, nullptr, nullptr, S, 0);
    final_kernel<<<BATCH, 256, 0, stream>>>(S, u, out);
}
